// RGAT_80496277062110
// MI455X (gfx1250) — matrix operand order repaired and checked by reading, not compiled, not run on hardware
//
#include <hip/hip_runtime.h>
#include <hip/hip_bf16.h>
#include <stdint.h>

#define N_NODES 50000
#define E_EDGES 800000
#define R_REL   3
#define L_LAYERS 2
#define H_HEADS 2
#define DH_DIM  128
#define HID     256
#define NCLS    64
#define BN_EPS  1e-5f
#define SLOPE   0.2f

typedef __attribute__((ext_vector_type(16))) __bf16      v16bf;
typedef __attribute__((ext_vector_type(8)))  float       v8f;
typedef __attribute__((ext_vector_type(8)))  unsigned int v8u;

// ---------------------------------------------------------------- helpers
__device__ __forceinline__ unsigned short f2bf(float f) {
    unsigned int u = __float_as_uint(f);
    u += 0x7FFFu + ((u >> 16) & 1u);          // round-to-nearest-even
    return (unsigned short)(u >> 16);
}

__device__ __forceinline__ void atomicMaxFloat(float* addr, float value) {
    // sign-flip trick: works with init pattern 0xFF800000 (-inf)
    if (value >= 0.f) atomicMax((int*)addr, __float_as_int(value));
    else              atomicMin((unsigned int*)addr, __float_as_uint(value));
}

// ---------------------------------------------------------------- f32 -> bf16
__global__ void k_f32_to_bf16(const float* __restrict__ in,
                              unsigned short* __restrict__ out, int n) {
    int t = blockIdx.x * blockDim.x + threadIdx.x;
    if (t < n) out[t] = f2bf(in[t]);
}

// transpose W[K][Ncol] f32 -> Wt[Ncol][K] bf16 (so GEMM B-loads are contiguous in K)
__global__ void k_transpose_bf16(const float* __restrict__ W,
                                 unsigned short* __restrict__ Wt,
                                 int K, int Ncol) {
    int t = blockIdx.x * blockDim.x + threadIdx.x;
    if (t >= K * Ncol) return;
    int k = t / Ncol, n = t % Ncol;
    Wt[(size_t)n * K + k] = f2bf(W[t]);
}

// ---------------------------------------------------------------- WMMA GEMM
// C[M x Ncol] = A[M x K](bf16) @ B[K x Ncol]   with B given transposed: Bt[Ncol][K] bf16
// One wave computes one 16x16 output tile; K stepped by 32 (v_wmma_f32_16x16x32_bf16).
__global__ void k_gemm_bf16(const unsigned short* __restrict__ A,
                            const unsigned short* __restrict__ Bt,
                            const float* __restrict__ bias,
                            float* __restrict__ C,
                            int M, int Ncol, int K, int has_bias) {
    const int lane  = threadIdx.x & 31;
    const int wave  = threadIdx.x >> 5;
    const int mtile = blockIdx.y * (blockDim.x >> 5) + wave;
    const int ntile = blockIdx.x;
    if (mtile * 16 >= M) return;                 // wave-uniform guard (EXEC all-1s for WMMA)

    const int r16  = lane & 15;                  // A row / B col / C col within tile
    const int half = lane >> 4;

    const unsigned short* arow = A  + (size_t)(mtile * 16 + r16) * K;
    const unsigned short* brow = Bt + (size_t)(ntile * 16 + r16) * K;

    v8f c = {};
    for (int kb = 0; kb < K; kb += 32) {
        // A fragment: lanes 0-15 hold K=kb+0..7 & kb+16..23, lanes 16-31 K=kb+8..15 & kb+24..31
        uint4 a0 = *(const uint4*)(arow + kb +      half * 8);
        uint4 a1 = *(const uint4*)(arow + kb + 16 + half * 8);
        v8u au; au[0]=a0.x; au[1]=a0.y; au[2]=a0.z; au[3]=a0.w;
                au[4]=a1.x; au[5]=a1.y; au[6]=a1.z; au[7]=a1.w;
        // B fragment: lanes 0-15 hold K=kb+0..7 & kb+16..23 of their column, lanes 16-31 K=kb+8..15 & kb+24..31
        uint4 b0 = *(const uint4*)(brow + kb +      half * 8);
        uint4 b1 = *(const uint4*)(brow + kb + 16 + half * 8);
        v8u bu; bu[0]=b0.x; bu[1]=b0.y; bu[2]=b0.z; bu[3]=b0.w;
                bu[4]=b1.x; bu[5]=b1.y; bu[6]=b1.z; bu[7]=b1.w;

        v16bf af = __builtin_bit_cast(v16bf, au);
        v16bf bfr = __builtin_bit_cast(v16bf, bu);
        c = __builtin_amdgcn_wmma_f32_16x16x32_bf16(
                false, af, false, bfr, (short)0, c, false, false);
    }

    const int col = ntile * 16 + r16;
    const float badd = has_bias ? bias[col] : 0.0f;
    #pragma unroll
    for (int v = 0; v < 8; ++v) {
        int row = mtile * 16 + v + half * 8;     // C layout: lanes16-31 -> M = v+8
        C[(size_t)row * Ncol + col] = c[v] + badd;
    }
}

// ---------------------------------------------------------------- el / er dots
// one wave per (node, head): el[g] = dot(z[n,h,:], al[h,:]), er likewise
__global__ void k_rowdot(const float* __restrict__ z,
                         const float* __restrict__ al,
                         const float* __restrict__ ar,
                         float* __restrict__ el, float* __restrict__ er) {
    int g = blockIdx.x * (blockDim.x >> 5) + (threadIdx.x >> 5);
    int lane = threadIdx.x & 31;
    if (g >= N_NODES * H_HEADS) return;
    int n = g / H_HEADS, h = g % H_HEADS;
    const float* zr  = z  + (size_t)n * HID + h * DH_DIM;
    const float* alr = al + h * DH_DIM;
    const float* arr = ar + h * DH_DIM;
    float sl = 0.f, sr = 0.f;
    #pragma unroll
    for (int d = lane; d < DH_DIM; d += 32) {
        float v = zr[d];
        sl += v * alr[d];
        sr += v * arr[d];
    }
    #pragma unroll
    for (int o = 16; o > 0; o >>= 1) {
        sl += __shfl_xor(sl, o, 32);
        sr += __shfl_xor(sr, o, 32);
    }
    if (lane == 0) { el[g] = sl; er[g] = sr; }
}

// ---------------------------------------------------------------- softmax init
__global__ void k_init_softmax(float* __restrict__ mx, float* __restrict__ sm, int n) {
    int t = blockIdx.x * blockDim.x + threadIdx.x;
    if (t < n) { mx[t] = __uint_as_float(0xFF800000u); sm[t] = 0.f; }
}

// per-(edge,head): leaky-relu(el[src]+er[dst]) -> eedge, atomic-max into mx[dst]
__global__ void k_edge_max(const int* __restrict__ src, const int* __restrict__ dst,
                           const float* __restrict__ el, const float* __restrict__ er,
                           float* __restrict__ eedge, float* __restrict__ mx, int E) {
    int t = blockIdx.x * blockDim.x + threadIdx.x;
    if (t >= E * H_HEADS) return;
    int e = t >> 1, h = t & 1;
    float v = el[src[e] * H_HEADS + h] + er[dst[e] * H_HEADS + h];
    v = v > 0.f ? v : SLOPE * v;
    eedge[t] = v;
    atomicMaxFloat(&mx[dst[e] * H_HEADS + h], v);
}

// per-(edge,head): ex = exp(e - mx[dst]) -> eedge, atomic-add into sm[dst]
__global__ void k_edge_exp(const int* __restrict__ dst, const float* __restrict__ mx,
                           float* __restrict__ eedge, float* __restrict__ sm, int E) {
    int t = blockIdx.x * blockDim.x + threadIdx.x;
    if (t >= E * H_HEADS) return;
    int e = t >> 1, h = t & 1;
    int d = dst[e];
    float ex = expf(eedge[t] - mx[d * H_HEADS + h]);
    eedge[t] = ex;
    atomicAdd(&sm[d * H_HEADS + h], ex);
}

// heavy scatter: 64 threads per edge (2 heads x 32 lanes x float4)
// h2[dst,h,d] += (ex/sm[dst,h]) * ew[e] * z[src,h,d]
__global__ void k_edge_scatter(const int* __restrict__ src, const int* __restrict__ dst,
                               const float* __restrict__ ew, const float* __restrict__ eedge,
                               const float* __restrict__ sm, const float* __restrict__ z,
                               float* __restrict__ h2, int E) {
    long long t = (long long)blockIdx.x * blockDim.x + threadIdx.x;
    if (t >= (long long)E * 64) return;
    int e = (int)(t >> 6);
    int k = (int)(t & 63);
    int h = k >> 5;
    int d = (k & 31) * 4;
    int s  = src[e];
    int dd = dst[e];
    float alpha = eedge[e * H_HEADS + h] * ew[e] / sm[dd * H_HEADS + h];
    float4 zv = *(const float4*)(z + (size_t)s * HID + h * DH_DIM + d);
    float* out = h2 + (size_t)dd * HID + h * DH_DIM + d;
    atomicAdd(out + 0, alpha * zv.x);
    atomicAdd(out + 1, alpha * zv.y);
    atomicAdd(out + 2, alpha * zv.z);
    atomicAdd(out + 3, alpha * zv.w);
}

// h2 init = sum over relations of bconv[l,r,:] broadcast per node
__global__ void k_init_h2(float* __restrict__ h2, const float* __restrict__ bconv_l) {
    int t = blockIdx.x * blockDim.x + threadIdx.x;
    if (t >= N_NODES * HID) return;
    int j = t & (HID - 1);
    h2[t] = bconv_l[j] + bconv_l[HID + j] + bconv_l[2 * HID + j];
}

// h = h1 + relu(h2)
__global__ void k_combine(const float* __restrict__ h1, const float* __restrict__ h2,
                          float* __restrict__ hout, int n) {
    int t = blockIdx.x * blockDim.x + threadIdx.x;
    if (t < n) hout[t] = h1[t] + fmaxf(h2[t], 0.f);
}

// bn(eval) -> relu -> bf16
__global__ void k_bn_relu_bf16(const float* __restrict__ h,
                               const float* __restrict__ gamma, const float* __restrict__ beta,
                               const float* __restrict__ rmean, const float* __restrict__ rvar,
                               unsigned short* __restrict__ out, int n) {
    int t = blockIdx.x * blockDim.x + threadIdx.x;
    if (t >= n) return;
    int j = t & (HID - 1);
    float v = (h[t] - rmean[j]) * rsqrtf(rvar[j] + BN_EPS) * gamma[j] + beta[j];
    out[t] = f2bf(fmaxf(v, 0.f));
}

// ---------------------------------------------------------------- launch
extern "C" void kernel_launch(void* const* d_in, const int* in_sizes, int n_in,
                              void* d_out, int out_size, void* d_ws, size_t ws_size,
                              hipStream_t stream) {
    (void)in_sizes; (void)n_in; (void)out_size; (void)ws_size;

    const float* x     = (const float*)d_in[0];
    const int*   src   = (const int*)  d_in[1];
    const int*   dst   = (const int*)  d_in[2];
    const float* ew    = (const float*)d_in[3];
    const float* Wfc   = (const float*)d_in[4];
    const float* al    = (const float*)d_in[5];
    const float* ar    = (const float*)d_in[6];
    const float* bconv = (const float*)d_in[7];
    const float* Wsk   = (const float*)d_in[8];
    const float* bsk   = (const float*)d_in[9];
    const float* gamma = (const float*)d_in[10];
    const float* beta  = (const float*)d_in[11];
    const float* rmean = (const float*)d_in[12];
    const float* rvar  = (const float*)d_in[13];
    const float* Wc    = (const float*)d_in[14];
    const float* bc    = (const float*)d_in[15];

    char* ws = (char*)d_ws;
    size_t off = 0;
    auto carve = [&](size_t bytes) -> void* {
        off = (off + 255) & ~(size_t)255;
        void* p = ws + off;
        off += bytes;
        return p;
    };
    float* hA = (float*)carve((size_t)N_NODES * HID * 4);      // current h
    float* hB = (float*)carve((size_t)N_NODES * HID * 4);      // skip result h1
    float* h2 = (float*)carve((size_t)N_NODES * HID * 4);      // GAT accumulator
    float* z  = (float*)carve((size_t)N_NODES * HID * 4);      // projection
    unsigned short* hbf  = (unsigned short*)carve((size_t)N_NODES * HID * 2);
    unsigned short* wskT = (unsigned short*)carve((size_t)L_LAYERS * HID * HID * 2);
    unsigned short* wfcT = (unsigned short*)carve((size_t)L_LAYERS * R_REL * HID * HID * 2);
    unsigned short* wcT  = (unsigned short*)carve((size_t)HID * NCLS * 2);
    float* el = (float*)carve((size_t)N_NODES * H_HEADS * 4);
    float* er = (float*)carve((size_t)N_NODES * H_HEADS * 4);
    float* mx = (float*)carve((size_t)N_NODES * H_HEADS * 4);
    float* sm = (float*)carve((size_t)N_NODES * H_HEADS * 4);
    float* eedge = (float*)carve((size_t)E_EDGES * H_HEADS * 4);

    const int NH   = N_NODES * HID;           // 12.8M
    const int MT   = N_NODES / 16;            // 3125 row tiles
    const dim3 gemmBlk(128);                  // 4 waves / block, 1 tile each

    // ---- weight prep: transpose + bf16 convert (tiny)
    {
        dim3 b(256), g((HID * HID + 255) / 256);
        for (int l = 0; l < L_LAYERS; ++l)
            k_transpose_bf16<<<g, b, 0, stream>>>(Wsk + (size_t)l * HID * HID,
                                                  wskT + (size_t)l * HID * HID, HID, HID);
        for (int l = 0; l < L_LAYERS; ++l)
            for (int r = 0; r < R_REL; ++r) {
                size_t o = (size_t)(l * R_REL + r) * HID * HID;
                k_transpose_bf16<<<g, b, 0, stream>>>(Wfc + o, wfcT + o, HID, HID);
            }
        k_transpose_bf16<<<dim3((HID * NCLS + 255) / 256), b, 0, stream>>>(Wc, wcT, HID, NCLS);
    }

    for (int l = 0; l < L_LAYERS; ++l) {
        const float* hin = (l == 0) ? x : hA;
        k_f32_to_bf16<<<(NH + 255) / 256, 256, 0, stream>>>(hin, hbf, NH);

        // skip path: hB = h @ Wsk[l] + bsk[l]
        dim3 gg(HID / 16, (MT + 3) / 4);
        k_gemm_bf16<<<gg, gemmBlk, 0, stream>>>(hbf, wskT + (size_t)l * HID * HID,
                                                bsk + (size_t)l * HID, hB,
                                                N_NODES, HID, HID, 1);
        // h2 = sum_r bconv[l,r] (broadcast)
        k_init_h2<<<(NH + 255) / 256, 256, 0, stream>>>(h2, bconv + (size_t)l * R_REL * HID);

        for (int r = 0; r < R_REL; ++r) {
            const int*   sr  = src + (size_t)r * E_EDGES;
            const int*   dr  = dst + (size_t)r * E_EDGES;
            const float* ewr = ew  + (size_t)r * E_EDGES;
            size_t wo = (size_t)(l * R_REL + r) * HID * HID;
            size_t ao = (size_t)(l * R_REL + r) * H_HEADS * DH_DIM;

            // z = h @ Wfc[l,r]
            k_gemm_bf16<<<gg, gemmBlk, 0, stream>>>(hbf, wfcT + wo, nullptr, z,
                                                    N_NODES, HID, HID, 0);
            // attention logits
            k_rowdot<<<(N_NODES * H_HEADS) / 8, 256, 0, stream>>>(z, al + ao, ar + ao, el, er);
            // edge softmax (grouped by dst) + weighted scatter
            k_init_softmax<<<(N_NODES * H_HEADS + 255) / 256, 256, 0, stream>>>(mx, sm, N_NODES * H_HEADS);
            k_edge_max<<<(E_EDGES * H_HEADS + 255) / 256, 256, 0, stream>>>(sr, dr, el, er, eedge, mx, E_EDGES);
            k_edge_exp<<<(E_EDGES * H_HEADS + 255) / 256, 256, 0, stream>>>(dr, mx, eedge, sm, E_EDGES);
            k_edge_scatter<<<(unsigned)(((long long)E_EDGES * 64 + 255) / 256), 256, 0, stream>>>(
                sr, dr, ewr, eedge, sm, z, h2, E_EDGES);
        }
        // h = h1 + relu(h2)   (in place into hA; all readers of old h done)
        k_combine<<<(NH + 255) / 256, 256, 0, stream>>>(hB, h2, hA, NH);
    }

    // classifier: bn -> relu -> bf16 -> GEMM into d_out
    k_bn_relu_bf16<<<(NH + 255) / 256, 256, 0, stream>>>(hA, gamma, beta, rmean, rvar, hbf, NH);
    dim3 gc(NCLS / 16, (MT + 3) / 4);
    k_gemm_bf16<<<gc, gemmBlk, 0, stream>>>(hbf, wcT, bc, (float*)d_out,
                                            N_NODES, NCLS, HID, 1);
}
